// FeaturePropagation_76192719831236
// MI455X (gfx1250) — matrix operand order repaired and checked by reading, not compiled, not run on hardware
//
#include <hip/hip_runtime.h>

// ---------------------------------------------------------------------------
// FeaturePropagation (PointNet++-style) for MI455X / gfx1250, wave32.
// GEMMs: v_wmma_f32_16x16x32_bf16, B tile staged through LDS (shared by all
// 4 waves of the block, double-buffered), 32x64 f32 accum tile per wave.
// ---------------------------------------------------------------------------

typedef __attribute__((ext_vector_type(16))) __bf16 v16bf;
typedef __attribute__((ext_vector_type(8)))  float  v8f;

#define Bb   4
#define Nn   8192
#define Ss   2048
#define D1c  128
#define D2c  256
#define Cc   256
#define KFUSE 384           // D1 + D2
#define NCOLS (Bb * Nn)     // 32768 points total

static __device__ __forceinline__ unsigned short f32_to_bf16(float f) {
    unsigned int u = __float_as_uint(f);
    unsigned int r = u + 0x7FFFu + ((u >> 16) & 1u);   // round-to-nearest-even
    return (unsigned short)(r >> 16);
}

// ---------------------------------------------------------------------------
// Pack weights W[M][K] (f32 row-major) into bf16 WMMA A-fragment order:
// frag(ktileIdx, mtileIdx)[lane][elem]; contiguous k order (matches B):
// lanes 0-15 -> K {0..15}, lanes 16-31 -> K {16..31}.
// ---------------------------------------------------------------------------
__global__ void pack_weights_kernel(const float* __restrict__ W,
                                    unsigned short* __restrict__ out,
                                    int M, int K) {
    int idx = blockIdx.x * 256 + threadIdx.x;
    int total = (M >> 4) * (K >> 5) * 512;
    if (idx >= total) return;
    int elem = idx & 15;
    int lane = (idx >> 4) & 31;
    int frag = idx >> 9;
    int Mtiles = M >> 4;
    int ktileIdx = frag / Mtiles;
    int mtileIdx = frag - ktileIdx * Mtiles;
    int row = (mtileIdx << 4) + (lane & 15);
    int kbase = (lane & 16) ? 16 : 0;
    int j = elem >> 1, h = elem & 1;
    int kk = elem; (void)j; (void)h;
    int k = (ktileIdx << 5) + kbase + kk;
    out[idx] = f32_to_bf16(W[row * K + k]);
}

// ---------------------------------------------------------------------------
// 3-NN inverse-distance interpolation + concat(points1, interp) + bf16 pack.
// Output X[col][KFUSE] bf16, col = b*N + n (WMMA B-operand layout).
// xyz2 (S=2048 pts) staged in LDS; one thread per query point.
// ---------------------------------------------------------------------------
__global__ void knn_interp_pack_kernel(const float* __restrict__ xyz1,
                                       const float* __restrict__ xyz2,
                                       const float* __restrict__ points1,
                                       const float* __restrict__ points2,
                                       unsigned short* __restrict__ X) {
    __shared__ float sx[Ss], sy[Ss], sz[Ss];
    int b = blockIdx.y;
    int tid = threadIdx.x;                       // 128 threads
    for (int i = tid; i < Ss; i += 128) {
        const float* p = xyz2 + ((size_t)b * Ss + i) * 3;
        sx[i] = p[0]; sy[i] = p[1]; sz[i] = p[2];
    }
    __syncthreads();

    int n = blockIdx.x * 128 + tid;
    const float* q = xyz1 + ((size_t)b * Nn + n) * 3;
    float qx = q[0], qy = q[1], qz = q[2];

    float d0 = 3.4e38f, d1 = 3.4e38f, d2 = 3.4e38f;
    int i0 = 0, i1 = 0, i2 = 0;
    for (int s = 0; s < Ss; ++s) {
        float dx = qx - sx[s], dy = qy - sy[s], dz = qz - sz[s];
        float d = dx * dx + dy * dy + dz * dz;
        if (d < d2) {
            if (d < d1) {
                if (d < d0) { d2 = d1; i2 = i1; d1 = d0; i1 = i0; d0 = d; i0 = s; }
                else        { d2 = d1; i2 = i1; d1 = d;  i1 = s; }
            } else          { d2 = d;  i2 = s; }
        }
    }
    float r0 = 1.f / (d0 + 1e-8f), r1 = 1.f / (d1 + 1e-8f), r2 = 1.f / (d2 + 1e-8f);
    float inv = 1.f / (r0 + r1 + r2);
    float w0 = r0 * inv, w1 = r1 * inv, w2 = r2 * inv;

    size_t col = (size_t)b * Nn + n;
    unsigned short* xo = X + col * KFUSE;

    const float* p1 = points1 + (size_t)b * D1c * Nn + n;      // [B,D1,N]
    for (int k = 0; k < D1c; ++k)
        xo[k] = f32_to_bf16(p1[(size_t)k * Nn]);

    const float* p2 = points2 + (size_t)b * D2c * Ss;          // [B,D2,S]
    for (int k = 0; k < D2c; ++k) {
        const float* row = p2 + (size_t)k * Ss;
        float v = w0 * row[i0] + w1 * row[i1] + w2 * row[i2];
        xo[D1c + k] = f32_to_bf16(v);
    }
}

// ---------------------------------------------------------------------------
// GEMM: Y[col][M] = W[M][K] @ X[K][col] + bias, via v_wmma_f32_16x16x32_bf16.
// Block = 4 waves (128 threads) -> 128(M) x 64(cols); per wave 32(M) x 64.
// B tile (64 cols x 32 K = 4 KB bf16) is staged into LDS once per K-step,
// shared by all 4 waves, double-buffered with one barrier per step.
// ---------------------------------------------------------------------------
template <int K>
__global__ __launch_bounds__(128)
void gemm_bias_kernel(const unsigned short* __restrict__ Apack,
                      const unsigned short* __restrict__ Bact,
                      const float* __restrict__ bias,
                      float* __restrict__ Y,
                      int M) {
    constexpr int KT = K / 32;
    __shared__ __align__(32) unsigned short Bt[2][64 * 32];  // [buf][col*32 + k]

    int tid  = threadIdx.x;
    int lane = tid & 31;
    int wave = tid >> 5;
    int m0 = blockIdx.y * 128 + wave * 32;      // this wave: rows m0 .. m0+31
    int colbase = blockIdx.x * 64;              // block cols colbase .. +63
    int Mtiles = M >> 4;
    int mt0 = m0 >> 4;

    const __bf16* Ap = (const __bf16*)Apack;
    const __bf16* Bp = (const __bf16*)Bact;

    // Cooperative staging: thread t moves 32 bytes (16 bf16) per K-step.
    int scol  = tid >> 1;                        // 0..63
    int shalf = (tid & 1) * 16;                  // 0 or 16 (bf16 elements)
    const __bf16* gB = Bp + (size_t)(colbase + scol) * K + shalf;
    auto stage = [&](int kt, int buf) {
        v16bf v = *(const v16bf*)(gB + kt * 32);
        *(v16bf*)((__bf16*)Bt[buf] + scol * 32 + shalf) = v;
    };

    int kbaseB = (lane & 16) ? 16 : 0;
    auto fragB = [&](int buf, int j) {
        int col = j * 16 + (lane & 15);
        return *(const v16bf*)((const __bf16*)Bt[buf] + col * 32 + kbaseB);
    };
    auto loadA = [&](int kt, int mt) {
        return *(const v16bf*)(Ap + (size_t)(kt * Mtiles + mt) * 512 + lane * 16);
    };

    v8f acc[2][4];
#pragma unroll
    for (int mi = 0; mi < 2; ++mi)
#pragma unroll
        for (int j = 0; j < 4; ++j)
            acc[mi][j] = (v8f){0.f,0.f,0.f,0.f,0.f,0.f,0.f,0.f};

    stage(0, 0);
    __syncthreads();

#pragma unroll
    for (int kt = 0; kt < KT; ++kt) {
        int cur = kt & 1, nxt = cur ^ 1;
        v16bf a0 = loadA(kt, mt0);
        v16bf a1 = loadA(kt, mt0 + 1);
        if (kt + 1 < KT) stage(kt + 1, nxt);     // overlaps WMMAs below
#pragma unroll
        for (int j = 0; j < 4; ++j) {
            v16bf bj = fragB(cur, j);
            acc[0][j] = __builtin_amdgcn_wmma_f32_16x16x32_bf16(
                false, a0, false, bj, (short)0, acc[0][j], false, false);
            acc[1][j] = __builtin_amdgcn_wmma_f32_16x16x32_bf16(
                false, a1, false, bj, (short)0, acc[1][j], false, false);
        }
        __syncthreads();   // cur fully read + nxt fully written before reuse
    }

#pragma unroll
    for (int mi = 0; mi < 2; ++mi) {
        int row0 = m0 + mi * 16 + ((lane & 16) ? 8 : 0);   // 8 consecutive chans
        v8f bias8 = *(const v8f*)(bias + row0);
#pragma unroll
        for (int j = 0; j < 4; ++j) {
            acc[mi][j] += bias8;
            int col = colbase + j * 16 + (lane & 15);
            *(v8f*)(Y + (size_t)col * M + row0) = acc[mi][j];
        }
    }
}

// ---------------------------------------------------------------------------
// BatchNorm partial stats over columns: per-channel sum / sumsq (atomics).
// Coalesced reads of Y[col][256].
// ---------------------------------------------------------------------------
__global__ void bn_stats_kernel(const float* __restrict__ Y,
                                float* __restrict__ sum, float* __restrict__ sq) {
    int c = threadIdx.x;                 // 256 channels
    int colbase = blockIdx.x * 128;
    float s = 0.f, q = 0.f;
    for (int j = 0; j < 128; ++j) {
        float v = Y[(size_t)(colbase + j) * Cc + c];
        s += v; q += v * v;
    }
    atomicAdd(&sum[c], s);
    atomicAdd(&sq[c], q);
}

__global__ void bn_finalize_kernel(const float* __restrict__ sum,
                                   const float* __restrict__ sq,
                                   const float* __restrict__ gamma,
                                   const float* __restrict__ beta,
                                   float* __restrict__ scale,
                                   float* __restrict__ shift) {
    int c = threadIdx.x;
    const float inv_n = 1.f / (float)NCOLS;
    float m = sum[c] * inv_n;
    float v = sq[c] * inv_n - m * m;
    float sc = gamma[c] * rsqrtf(v + 1e-5f);
    scale[c] = sc;
    shift[c] = beta[c] - m * sc;
}

// Apply BN + ReLU and pack bf16 activations [col][256] for the next GEMM.
__global__ void bn_apply_pack_kernel(const float* __restrict__ Y,
                                     const float* __restrict__ scale,
                                     const float* __restrict__ shift,
                                     unsigned short* __restrict__ Xn) {
    int idx = blockIdx.x * 256 + threadIdx.x;
    int c = idx & (Cc - 1);
    float v = fmaxf(Y[idx] * scale[c] + shift[c], 0.f);
    Xn[idx] = f32_to_bf16(v);
}

// ---------------------------------------------------------------------------
// Final: out[b][c][n] = relu( relu(BN0(Y1)) + BN2(Y3) ).
// LDS 32x32 tile transpose: coalesced reads of [col][C], coalesced writes
// of the channel-major [B, C, N] output.
// ---------------------------------------------------------------------------
__global__ void final_out_kernel(const float* __restrict__ Y1,
                                 const float* __restrict__ Y3,
                                 const float* __restrict__ s0,
                                 const float* __restrict__ t0,
                                 const float* __restrict__ s2,
                                 const float* __restrict__ t2,
                                 float* __restrict__ out) {
    __shared__ float tile[32][33];
    int tx = threadIdx.x;                 // 32
    int ty = threadIdx.y;                 // 8
    int coltile = blockIdx.x * 32;
    int ctile = blockIdx.y * 32;
    for (int yy = 0; yy < 4; ++yy) {
        int c = ctile + tx;
        int col = coltile + ty + yy * 8;
        size_t idx = (size_t)col * Cc + c;
        float x = fmaxf(Y1[idx] * s0[c] + t0[c], 0.f);
        float y = Y3[idx] * s2[c] + t2[c];
        tile[ty + yy * 8][tx] = fmaxf(x + y, 0.f);
    }
    __syncthreads();
    int b = coltile >> 13;                // 8192 cols per batch
    int n0 = coltile & (Nn - 1);
    for (int yy = 0; yy < 4; ++yy) {
        int c = ctile + ty + yy * 8;
        out[(size_t)b * Cc * Nn + (size_t)c * Nn + n0 + tx] = tile[tx][ty + yy * 8];
    }
}

// ---------------------------------------------------------------------------
extern "C" void kernel_launch(void* const* d_in, const int* in_sizes, int n_in,
                              void* d_out, int out_size, void* d_ws, size_t ws_size,
                              hipStream_t stream) {
    const float* xyz1    = (const float*)d_in[0];
    const float* xyz2    = (const float*)d_in[1];
    const float* points1 = (const float*)d_in[2];
    const float* points2 = (const float*)d_in[3];
    const float* W_fuse  = (const float*)d_in[4];
    const float* b_fuse  = (const float*)d_in[5];
    const float* g_fuse  = (const float*)d_in[6];
    const float* be_fuse = (const float*)d_in[7];
    const float* W1      = (const float*)d_in[8];
    const float* b1      = (const float*)d_in[9];
    const float* g1      = (const float*)d_in[10];
    const float* be1     = (const float*)d_in[11];
    const float* W2      = (const float*)d_in[12];
    const float* b2      = (const float*)d_in[13];
    const float* g2      = (const float*)d_in[14];
    const float* be2     = (const float*)d_in[15];
    float* out = (float*)d_out;

    char* ws = (char*)d_ws;
    // Workspace layout (buffers aliased across lifetimes):
    size_t o_Xfuse = 0;                                   // 32768*384*2  = 25.2 MB
    size_t o_Y1    = o_Xfuse + (size_t)NCOLS * KFUSE * 2; // 32768*256*4  = 33.6 MB
    size_t o_Xact  = o_Y1    + (size_t)NCOLS * Cc * 4;    // 32768*256*2  = 16.8 MB (reused)
    size_t o_Y23   = o_Xact  + (size_t)NCOLS * Cc * 2;    // 32768*256*4  = 33.6 MB (Y2 then Y3)
    size_t o_Wf    = o_Y23   + (size_t)NCOLS * Cc * 4;    // 12*16*512*2
    size_t o_W1    = o_Wf    + (size_t)12 * 16 * 512 * 2; //  8*16*512*2
    size_t o_W2    = o_W1    + (size_t)8  * 16 * 512 * 2;
    size_t o_stat  = o_W2    + (size_t)8  * 16 * 512 * 2; // 3 stages * 4KB

    unsigned short* Xfuse = (unsigned short*)(ws + o_Xfuse);
    float*          Y1    = (float*)(ws + o_Y1);
    unsigned short* Xact  = (unsigned short*)(ws + o_Xact);
    float*          Y23   = (float*)(ws + o_Y23);
    unsigned short* WfP   = (unsigned short*)(ws + o_Wf);
    unsigned short* W1P   = (unsigned short*)(ws + o_W1);
    unsigned short* W2P   = (unsigned short*)(ws + o_W2);
    float* st = (float*)(ws + o_stat);
    float *sum0 = st,        *sq0 = st + 256,  *sc0 = st + 512,  *sh0 = st + 768;
    float *sum1 = st + 1024, *sq1 = st + 1280, *sc1 = st + 1536, *sh1 = st + 1792;
    float *sum2 = st + 2048, *sq2 = st + 2304, *sc2 = st + 2560, *sh2 = st + 2816;

    hipMemsetAsync(st, 0, 3 * 4096, stream);

    // Weight repack (bf16, WMMA fragment order)
    pack_weights_kernel<<<(12 * 16 * 512) / 256, 256, 0, stream>>>(W_fuse, WfP, Cc, KFUSE);
    pack_weights_kernel<<<( 8 * 16 * 512) / 256, 256, 0, stream>>>(W1, W1P, Cc, Cc);
    pack_weights_kernel<<<( 8 * 16 * 512) / 256, 256, 0, stream>>>(W2, W2P, Cc, Cc);

    // 3-NN interpolation + concat + bf16 pack
    knn_interp_pack_kernel<<<dim3(Nn / 128, Bb), 128, 0, stream>>>(
        xyz1, xyz2, points1, points2, Xfuse);

    // fuse: conv + BN + ReLU
    gemm_bias_kernel<KFUSE><<<dim3(NCOLS / 64, Cc / 128), 128, 0, stream>>>(
        WfP, Xfuse, b_fuse, Y1, Cc);
    bn_stats_kernel<<<NCOLS / 128, 256, 0, stream>>>(Y1, sum0, sq0);
    bn_finalize_kernel<<<1, 256, 0, stream>>>(sum0, sq0, g_fuse, be_fuse, sc0, sh0);
    bn_apply_pack_kernel<<<(NCOLS * Cc) / 256, 256, 0, stream>>>(Y1, sc0, sh0, Xact);

    // conv1 + BN + ReLU  (Y2 in Y23)
    gemm_bias_kernel<Cc><<<dim3(NCOLS / 64, Cc / 128), 128, 0, stream>>>(
        W1P, Xact, b1, Y23, Cc);
    bn_stats_kernel<<<NCOLS / 128, 256, 0, stream>>>(Y23, sum1, sq1);
    bn_finalize_kernel<<<1, 256, 0, stream>>>(sum1, sq1, g1, be1, sc1, sh1);
    bn_apply_pack_kernel<<<(NCOLS * Cc) / 256, 256, 0, stream>>>(Y23, sc1, sh1, Xact);

    // conv2 + BN  (Y3 overwrites Y23; Y2 fully consumed above)
    gemm_bias_kernel<Cc><<<dim3(NCOLS / 64, Cc / 128), 128, 0, stream>>>(
        W2P, Xact, b2, Y23, Cc);
    bn_stats_kernel<<<NCOLS / 128, 256, 0, stream>>>(Y23, sum2, sq2);
    bn_finalize_kernel<<<1, 256, 0, stream>>>(sum2, sq2, g2, be2, sc2, sh2);

    // residual add + ReLU, transpose to [B, C, N]
    final_out_kernel<<<dim3(NCOLS / 32, Cc / 32), dim3(32, 8), 0, stream>>>(
        Y1, Y23, sc0, sh0, sc2, sh2, out);
}
